// Conv2D_wd_29128468201846
// MI455X (gfx1250) — matrix operand order repaired and checked by reading, not compiled, not run on hardware
//
#include <hip/hip_runtime.h>

// ---------------------------------------------------------------------------
// MI455X (gfx1250) masked conv2d:
//   outputs[b,h,w,f] = sum_{kh,kw,c} in[b,h+kh-1,w+kw-1,c] * (kernel*mask_c)[b,kh,kw,c,f]
//                      + (bias*mask_b)[b,f]
// Strategy: precompute masked weights into WMMA B-fragment layout (f16),
// then implicit-GEMM with v_wmma_f32_16x16x32_f16, f32 accumulation.
// ---------------------------------------------------------------------------

typedef __attribute__((ext_vector_type(16))) _Float16 v16h;
typedef __attribute__((ext_vector_type(8)))  float    v8f;
typedef __attribute__((ext_vector_type(4)))  float    v4f;

#define CD5_B   8
#define CD5_H   512
#define CD5_W   512
#define CD5_C   32     // Cin
#define CD5_F   32     // filters
#define CD5_T   9      // 3x3 taps
// B-fragment storage: [B][tap][nh(0..1)][lane(0..31)][elem(0..15)] f16
#define WB_HALVES_PER_TILE 512                      // 32 lanes * 16 halves
#define WB_BYTES (CD5_B * CD5_T * 2 * WB_HALVES_PER_TILE * 2)  // 147456 B

__device__ __forceinline__ float gumbel_keep(float u) {
    // mask = 1 - sigmoid((log(p/(1-p)) + log(u+eps) - log(1-u+eps)) / 0.01)
    // p = 0.2 for both mask_c (sigmoid(log(.25)) == .2) and mask_b  =>
    // constant term log(0.2/0.8) = log(0.25)
    const float EPS = 1e-30f;
    float logit = -1.3862943611198906f + logf(u + EPS) - logf(1.0f - u + EPS);
    float x = logit * 100.0f;                 // / TEMP(0.01)
    float s = 1.0f / (1.0f + expf(-x));       // saturates cleanly in f32
    return 1.0f - s;
}

// ---- masked kernel -> f16 WMMA B-fragment layout ---------------------------
// B tile is 32x16 (K=cin x N=f-half). ISA 16-bit B layout (wave32):
//   lanes 0-15  hold K=0..7,16..23  (elem s: K=s for s<8, K=8+s for s>=8),  column N = lane
//   lanes 16-31 hold K=8..15,24..31 (elem s: K=8+s for s<8, K=16+s for s>=8), column N = lane-16
// => lane = (f&15) + 16*((cin>>3)&1), elem = (cin&7) + 8*(cin>=16)
__global__ void cd5_mask_weights(const float* __restrict__ kernel,
                                 const float* __restrict__ unif_c,
                                 _Float16* __restrict__ wB) {
    int idx = blockIdx.x * 256 + threadIdx.x;           // B*9*32*32 = 73728
    if (idx >= CD5_B * CD5_T * CD5_C * CD5_F) return;
    int f   = idx & 31;
    int cin = (idx >> 5) & 31;
    int t   = (idx >> 10) % CD5_T;
    int b   = idx / (CD5_T * CD5_C * CD5_F);
    // unif_c flat layout [B,3,3,32,32] == ((b*9+t)*32+cin)*32+f == idx
    float w = kernel[(t * CD5_C + cin) * CD5_F + f] * gumbel_keep(unif_c[idx]);
    int nh   = f >> 4;
    int lane = (f & 15) + ((cin & 8) ? 16 : 0);
    int e    = (cin & 7) + ((cin & 16) ? 8 : 0);
    wB[((((b * CD5_T + t) * 2 + nh) * 32 + lane) << 4) + e] = (_Float16)w;
}

__global__ void cd5_mask_bias(const float* __restrict__ bias,
                              const float* __restrict__ unif_b,
                              float* __restrict__ mb) {
    int idx = threadIdx.x;                               // B*F = 256
    mb[idx] = bias[idx & 31] * gumbel_keep(unif_b[idx]);
}

// ---- main conv: implicit GEMM, one 16-pixel x 32-filter tile per wave ------
__global__ __launch_bounds__(256)
void cd5_conv_wmma(const float* __restrict__ inp,
                   const _Float16* __restrict__ wB,
                   const float* __restrict__ mb,
                   float* __restrict__ out) {
    const int lane = threadIdx.x & 31;
    const int wid  = threadIdx.x >> 5;        // 8 waves per workgroup
    const int b    = blockIdx.z;
    const int h    = blockIdx.y;
    const int w0   = blockIdx.x * 128 + wid * 16;   // 16 consecutive pixels
    const int m    = lane & 15;               // pixel row of A handled by lane
    // A-fragment channel groups: lane<16 -> ch {0..7,16..23}; lane>=16 -> {8..15,24..31}
    const int cbase = (lane < 16) ? 0 : 8;

    v8f acc0 = {};   // filters 0..15
    v8f acc1 = {};   // filters 16..31

    const _Float16* wBb = wB + (size_t)b * (CD5_T * 2 * WB_HALVES_PER_TILE);

#pragma unroll
    for (int kh = 0; kh < 3; ++kh) {
        const int hy = h + kh - 1;
        const bool rowok = ((unsigned)hy < (unsigned)CD5_H);
#pragma unroll
        for (int kw = 0; kw < 3; ++kw) {
            const int t  = kh * 3 + kw;
            const int wx = w0 + m + kw - 1;
            v16h a = {};
            if (rowok && ((unsigned)wx < (unsigned)CD5_W)) {
                const float* p = inp +
                    (((size_t)b * CD5_H + (size_t)hy) * CD5_W + (size_t)wx) * CD5_C + cbase;
                v4f x0 = *(const v4f*)(p);        // ch cbase+0..3
                v4f x1 = *(const v4f*)(p + 4);    // ch cbase+4..7
                v4f x2 = *(const v4f*)(p + 16);   // ch cbase+16..19
                v4f x3 = *(const v4f*)(p + 20);   // ch cbase+20..23
                a[0]  = (_Float16)x0[0]; a[1]  = (_Float16)x0[1];
                a[2]  = (_Float16)x0[2]; a[3]  = (_Float16)x0[3];
                a[4]  = (_Float16)x1[0]; a[5]  = (_Float16)x1[1];
                a[6]  = (_Float16)x1[2]; a[7]  = (_Float16)x1[3];
                a[8]  = (_Float16)x2[0]; a[9]  = (_Float16)x2[1];
                a[10] = (_Float16)x2[2]; a[11] = (_Float16)x2[3];
                a[12] = (_Float16)x3[0]; a[13] = (_Float16)x3[1];
                a[14] = (_Float16)x3[2]; a[15] = (_Float16)x3[3];
            }
            const _Float16* pb = wBb + (size_t)t * (2 * WB_HALVES_PER_TILE) + lane * 16;
            v16h b0 = *(const v16h*)(pb);
            v16h b1 = *(const v16h*)(pb + WB_HALVES_PER_TILE);
            // D = A(16x32 f16) x B(32x16 f16) + C(f32)
            acc0 = __builtin_amdgcn_wmma_f32_16x16x32_f16(
                       false, a, false, b0, (short)0, acc0, false, false);
            acc1 = __builtin_amdgcn_wmma_f32_16x16x32_f16(
                       false, a, false, b1, (short)0, acc1, false, false);
        }
    }

    // C/D layout: VGPR r -> M = r + 8*(lane>=16), N = lane&15
    const float bias0 = mb[b * CD5_F + m];
    const float bias1 = mb[b * CD5_F + 16 + m];
    const int mrow = (lane >> 4) * 8;
#pragma unroll
    for (int r = 0; r < 8; ++r) {
        const int pix = w0 + mrow + r;
        size_t o = (((size_t)b * CD5_H + (size_t)h) * CD5_W + (size_t)pix) * CD5_F;
        out[o + m]      = acc0[r] + bias0;
        out[o + 16 + m] = acc1[r] + bias1;
    }
}

extern "C" void kernel_launch(void* const* d_in, const int* in_sizes, int n_in,
                              void* d_out, int out_size, void* d_ws, size_t ws_size,
                              hipStream_t stream) {
    const float* inp    = (const float*)d_in[0];   // [8,512,512,32]
    const float* kernel = (const float*)d_in[1];   // [3,3,32,32]
    const float* bias   = (const float*)d_in[2];   // [32]
    const float* unif_c = (const float*)d_in[3];   // [8,3,3,32,32]
    const float* unif_b = (const float*)d_in[4];   // [8,32]

    _Float16* wB = (_Float16*)d_ws;                        // 147456 B
    float*    mb = (float*)((char*)d_ws + WB_BYTES);       // 1024 B

    const int nW = CD5_B * CD5_T * CD5_C * CD5_F;          // 73728
    cd5_mask_weights<<<(nW + 255) / 256, 256, 0, stream>>>(kernel, unif_c, wB);
    cd5_mask_bias<<<1, CD5_B * CD5_F, 0, stream>>>(bias, unif_b, mb);

    dim3 grid(CD5_W / 128, CD5_H, CD5_B);                  // (4, 512, 8)
    cd5_conv_wmma<<<grid, 256, 0, stream>>>(inp, wB, mb, (float*)d_out);
}
